// HybridSSM_90486370992526
// MI455X (gfx1250) — matrix operand order repaired and checked by reading, not compiled, not run on hardware
//
#include <hip/hip_runtime.h>
#include <hip/hip_bf16.h>
#include <math.h>

typedef __attribute__((ext_vector_type(16))) __bf16 v16bf;
typedef __attribute__((ext_vector_type(8)))  __bf16 v8bf;
typedef __attribute__((ext_vector_type(8)))  float  v8f;

#define NN 8
#define VV 1862
#define CC 64
#define TT 150
#define NVTOT (NN*VV)          // 14896
#define EE 238336
#define NCAT 1792              // 27*64 (spline) + 64 (root)
#define NROWTILES (NVTOT/16)   // 931

__device__ __forceinline__ float fast_tanh(float x) {
#if __has_builtin(__builtin_amdgcn_tanhf)
  return __builtin_amdgcn_tanhf(x);    // v_tanh_f32 (gfx1250 TRANS op)
#else
  return tanhf(x);
#endif
}

__device__ __forceinline__ v8f wmma_bf16(v16bf a, v16bf b, v8f c) {
  // D = A(16x32 bf16) x B(32x16 bf16) + C(16x16 f32)
  return __builtin_amdgcn_wmma_f32_16x16x32_bf16(false, a, false, b, (short)0, c, false, false);
}

// A operand: lane holds row m = lane%16. Elements j<8 -> K = 32*kc + 8*half + j,
// j>=8 -> K = 32*kc + 16 + 8*half + (j-8). Two contiguous 16B loads per lane.
__device__ __forceinline__ v16bf load_a_row(const __bf16* row, int kc, int half) {
  v8bf lo = *(const v8bf*)(row + kc*32 + half*8);
  v8bf hi = *(const v8bf*)(row + kc*32 + 16 + half*8);
  v16bf r;
#pragma unroll
  for (int i = 0; i < 8; i++) { r[i] = lo[i]; r[i+8] = hi[i]; }
  return r;
}

// B operand from transposed weights Wt[n][k] (row = output column, 64 k's each):
// lane holds column n = lane%16; element j<8 -> K = 32*kc + 8*half + j,
// j>=8 -> K = 32*kc + 16 + 8*half + (j-8). Two contiguous 16B loads per lane.
__device__ __forceinline__ v16bf load_b_t(const __bf16* Wt, int ncol, int kc, int half) {
  return load_a_row(Wt + ncol*64, kc, half);
}

// ---------------- setup kernels (once per launch) ----------------

__global__ void k_weights(const float* __restrict__ Wspl, const float* __restrict__ Wroot,
                          const float* __restrict__ Wx, const float* __restrict__ Wo1,
                          const float* __restrict__ Wo2,
                          __bf16* __restrict__ Wt1, __bf16* __restrict__ Wt2,
                          __bf16* __restrict__ Wtx, __bf16* __restrict__ Wcat) {
  int i = blockIdx.x * 256 + threadIdx.x;
  if (i < 64*64) {
    int n = i >> 6, c = i & 63;
    Wt1[n*64 + c] = (__bf16)Wo1[c*64 + n];
    Wt2[n*64 + c] = (__bf16)Wo2[c*64 + n];
    Wtx[n*64 + c] = (__bf16)Wx [c*64 + n];
  }
  if (i < NCAT*64) {
    int n = i >> 6, c = i & 63;
    float v;
    if (n < 1728) {
      int kk = n / 64, d = n % 64;
      v = Wspl[(kk*64 + c)*64 + d];          // W_spline[kk][c][d]
    } else {
      v = Wroot[c*64 + (n - 1728)];
    }
    Wcat[n*64 + c] = (__bf16)v;
  }
}

__global__ void k_h0(const float* __restrict__ x, float* __restrict__ h,
                     __bf16* __restrict__ hbf, float* __restrict__ out,
                     float* __restrict__ agg) {
  int i = blockIdx.x * 256 + threadIdx.x;      // i = nv*64 + c
  if (i >= NVTOT*64) return;
  float v = x[(size_t)i * TT];                 // x[..., t=0]
  h[i] = v; hbf[i] = (__bf16)v;
  out[(size_t)i * TT] = v;                     // out[..., t=0] = h0
  agg[i] = 0.f;
}

__global__ void k_edges(const int* __restrict__ eidx, const float* __restrict__ eattr,
                        unsigned int* __restrict__ eoff, float* __restrict__ ecoef) {
  int e = blockIdx.x * 256 + threadIdx.x;
  if (e >= EE) return;
  int src = eidx[e];
  float fr[3]; int lo[3];
#pragma unroll
  for (int d = 0; d < 3; d++) {
    float p = fminf(fmaxf(eattr[e*3 + d], 0.f), 1.f) * 2.0f;   // * (K-1)
    float fl = fminf(fmaxf(floorf(p), 0.f), 1.0f);             // clip to K-2
    lo[d] = (int)fl; fr[d] = p - fl;
  }
#pragma unroll
  for (int b = 0; b < 8; b++) {
    int idx = 0, stride = 1; float coeff = 1.f;
#pragma unroll
    for (int d = 0; d < 3; d++) {
      int bit = (b >> d) & 1;
      idx += (lo[d] + bit) * stride;
      stride *= 3;
      coeff *= bit ? fr[d] : (1.f - fr[d]);
    }
    eoff [e*8 + b] = (unsigned)((src*27 + idx) * 64 * 2);  // byte offset into bf16 y_all
    ecoef[e*8 + b] = coeff;
  }
}

// ---------------- per-step kernels ----------------

// h_ode = h + tanh(h @ W1) @ W2 ; writes f32 + bf16 mirrors
__global__ void k_ode(const float* __restrict__ h, const __bf16* __restrict__ hbf,
                      const __bf16* __restrict__ Wt1, const __bf16* __restrict__ Wt2,
                      float* __restrict__ hode, __bf16* __restrict__ hodebf) {
  __shared__ __bf16 lds[8][16 * 80];           // per-wave 16x64 tile, stride 80 (160B, 16B-mult)
  int w = threadIdx.x >> 5;
  int tile = blockIdx.x * 8 + w;
  if (tile >= NROWTILES) return;
  int L = threadIdx.x & 31, n = L & 15, half = L >> 4;
  int m0 = tile * 16;

  const __bf16* arow = hbf + (m0 + n) * 64;
  v16bf a0 = load_a_row(arow, 0, half);
  v16bf a1 = load_a_row(arow, 1, half);
  __bf16* my = lds[w];

#pragma unroll
  for (int nt = 0; nt < 4; nt++) {
    v8f acc = {};
    acc = wmma_bf16(a0, load_b_t(Wt1, nt*16 + n, 0, half), acc);
    acc = wmma_bf16(a1, load_b_t(Wt1, nt*16 + n, 1, half), acc);
#pragma unroll
    for (int r = 0; r < 8; r++)                // C layout: lane n, row m = 8*half + r
      my[(8*half + r) * 80 + nt*16 + n] = (__bf16)fast_tanh(acc[r]);
  }
  asm volatile("s_wait_dscnt 0x0" ::: "memory");

  const __bf16* arow2 = my + n * 80;           // transposed tile: lane's A row
  v16bf c0 = load_a_row(arow2, 0, half);
  v16bf c1 = load_a_row(arow2, 1, half);
#pragma unroll
  for (int nt = 0; nt < 4; nt++) {
    v8f acc = {};
    acc = wmma_bf16(c0, load_b_t(Wt2, nt*16 + n, 0, half), acc);
    acc = wmma_bf16(c1, load_b_t(Wt2, nt*16 + n, 1, half), acc);
#pragma unroll
    for (int r = 0; r < 8; r++) {
      int idx = (m0 + 8*half + r) * 64 + nt*16 + n;
      float v = h[idx] + acc[r];
      hode[idx] = v; hodebf[idx] = (__bf16)v;
    }
  }
}

// y_all[nv][27*64] (bf16) and R = h_ode@W_root (f32) via one 64x1792 GEMM
__global__ void k_spline(const __bf16* __restrict__ hodebf, const __bf16* __restrict__ Wcat,
                         __bf16* __restrict__ yall, float* __restrict__ R) {
  int w = threadIdx.x >> 5;
  int id = blockIdx.x * 8 + w;                 // 931 row-tiles x 4 col-chunks
  if (id >= NROWTILES * 4) return;
  int tile = id >> 2, chunk = id & 3;          // 28 n-tiles per chunk (112 total)
  int L = threadIdx.x & 31, n = L & 15, half = L >> 4;
  int m0 = tile * 16;

  const __bf16* arow = hodebf + (m0 + n) * 64;
  v16bf a0 = load_a_row(arow, 0, half);
  v16bf a1 = load_a_row(arow, 1, half);

#pragma unroll 4
  for (int nt = chunk * 28; nt < chunk * 28 + 28; nt++) {
    v8f acc = {};
    acc = wmma_bf16(a0, load_b_t(Wcat, nt*16 + n, 0, half), acc);
    acc = wmma_bf16(a1, load_b_t(Wcat, nt*16 + n, 1, half), acc);
    if (nt < 108) {
#pragma unroll
      for (int r = 0; r < 8; r++)
        yall[(size_t)(m0 + 8*half + r) * 1728 + nt*16 + n] = (__bf16)acc[r];
    } else {
#pragma unroll
      for (int r = 0; r < 8; r++)
        R[(m0 + 8*half + r) * 64 + (nt - 108)*16 + n] = acc[r];
    }
  }
}

// one wave per edge: 8 corner gathers from L2-resident bf16 y_all, fp32 atomic scatter
__global__ void k_gather(const unsigned char* __restrict__ ybytes,
                         const unsigned int* __restrict__ eoff,
                         const float* __restrict__ ecoef,
                         const int* __restrict__ eidx,
                         float* __restrict__ agg) {
  int e = blockIdx.x * 8 + (threadIdx.x >> 5);
  if (e >= EE) return;
  int L = threadIdx.x & 31;
  int dst = eidx[EE + e];
  float acc0 = 0.f, acc1 = 0.f;
#pragma unroll
  for (int b = 0; b < 8; b++) {
    unsigned int off = eoff[e*8 + b];          // wave-uniform -> SMEM load
    float c = ecoef[e*8 + b];
    unsigned int u = *(const unsigned int*)(ybytes + off + L*4);   // 2 bf16 / lane
    acc0 = fmaf(c, __uint_as_float(u << 16), acc0);
    acc1 = fmaf(c, __uint_as_float(u & 0xffff0000u), acc1);
  }
  unsafeAtomicAdd(&agg[dst*64 + 2*L],     acc0);
  unsafeAtomicAdd(&agg[dst*64 + 2*L + 1], acc1);
}

// h = tanh(agg + R + x_t@W_x + b); writes out[...,t]; clears agg for next step
__global__ void k_update(const float* __restrict__ x, int t,
                         const __bf16* __restrict__ Wtx,
                         const float* __restrict__ R, float* __restrict__ agg,
                         const float* __restrict__ bias,
                         float* __restrict__ h, __bf16* __restrict__ hbf,
                         float* __restrict__ out) {
  int w = threadIdx.x >> 5;
  int tile = blockIdx.x * 8 + w;
  if (tile >= NROWTILES) return;
  int L = threadIdx.x & 31, n = L & 15, half = L >> 4;
  int m0 = tile * 16;

  // build bf16 A rows of x_t from strided f32 loads (stride T)
  const float* xr = x + (size_t)(m0 + n) * 64 * TT + t;
  v16bf a0, a1;
#pragma unroll
  for (int j = 0; j < 8; j++) {
    a0[j]     = (__bf16)xr[(size_t)(      half*8 + j) * TT];
    a0[j + 8] = (__bf16)xr[(size_t)(16 +  half*8 + j) * TT];
    a1[j]     = (__bf16)xr[(size_t)(32 +  half*8 + j) * TT];
    a1[j + 8] = (__bf16)xr[(size_t)(48 +  half*8 + j) * TT];
  }
#pragma unroll
  for (int nt = 0; nt < 4; nt++) {
    v8f acc = {};
    acc = wmma_bf16(a0, load_b_t(Wtx, nt*16 + n, 0, half), acc);
    acc = wmma_bf16(a1, load_b_t(Wtx, nt*16 + n, 1, half), acc);
    int col = nt*16 + n;
    float bb = bias[col];
#pragma unroll
    for (int r = 0; r < 8; r++) {
      int idx = (m0 + 8*half + r) * 64 + col;
      float v = fast_tanh(acc[r] + agg[idx] + R[idx] + bb);
      agg[idx] = 0.f;                          // ready for next step's atomics
      h[idx] = v; hbf[idx] = (__bf16)v;
      out[(size_t)idx * TT + t] = v;           // out[(nv*C+c)*T + t]
    }
  }
}

extern "C" void kernel_launch(void* const* d_in, const int* in_sizes, int n_in,
                              void* d_out, int out_size, void* d_ws, size_t ws_size,
                              hipStream_t stream) {
  const float* x     = (const float*)d_in[0];
  const int*   eidx  = (const int*)  d_in[1];
  const float* eattr = (const float*)d_in[2];
  const float* Wspl  = (const float*)d_in[3];
  const float* Wroot = (const float*)d_in[4];
  const float* Wx    = (const float*)d_in[5];
  const float* bias  = (const float*)d_in[6];
  const float* Wo1   = (const float*)d_in[7];
  const float* Wo2   = (const float*)d_in[8];
  float* out = (float*)d_out;
  (void)in_sizes; (void)n_in; (void)out_size; (void)ws_size;

  char* p = (char*)d_ws;
  auto alloc = [&](size_t bytes) -> char* {
    char* r = p; p += (bytes + 255) & ~(size_t)255; return r;
  };
  float*        h      = (float*)       alloc((size_t)NVTOT*64*4);
  __bf16*       hbf    = (__bf16*)      alloc((size_t)NVTOT*64*2);
  float*        hode   = (float*)       alloc((size_t)NVTOT*64*4);
  __bf16*       hodebf = (__bf16*)      alloc((size_t)NVTOT*64*2);
  float*        R      = (float*)       alloc((size_t)NVTOT*64*4);
  float*        agg    = (float*)       alloc((size_t)NVTOT*64*4);
  __bf16*       yall   = (__bf16*)      alloc((size_t)NVTOT*1728*2);
  __bf16*       Wt1    = (__bf16*)      alloc(64*64*2);
  __bf16*       Wt2    = (__bf16*)      alloc(64*64*2);
  __bf16*       Wtx    = (__bf16*)      alloc(64*64*2);
  __bf16*       Wcat   = (__bf16*)      alloc((size_t)NCAT*64*2);
  unsigned int* eoff   = (unsigned int*)alloc((size_t)EE*8*4);
  float*        ecoef  = (float*)       alloc((size_t)EE*8*4);

  k_weights<<<(NCAT*64 + 255)/256, 256, 0, stream>>>(Wspl, Wroot, Wx, Wo1, Wo2,
                                                     Wt1, Wt2, Wtx, Wcat);
  k_h0<<<(NVTOT*64 + 255)/256, 256, 0, stream>>>(x, h, hbf, out, agg);
  k_edges<<<(EE + 255)/256, 256, 0, stream>>>(eidx, eattr, eoff, ecoef);

  for (int t = 1; t < TT; t++) {
    k_ode   <<<(NROWTILES + 7)/8,   256, 0, stream>>>(h, hbf, Wt1, Wt2, hode, hodebf);
    k_spline<<<(NROWTILES*4 + 7)/8, 256, 0, stream>>>(hodebf, Wcat, yall, R);
    k_gather<<<EE/8,                256, 0, stream>>>((const unsigned char*)yall,
                                                      eoff, ecoef, eidx, agg);
    k_update<<<(NROWTILES + 7)/8,   256, 0, stream>>>(x, t, Wtx, R, agg, bias,
                                                      h, hbf, out);
  }
}
